// DHSpaceKAA_57827439673618
// MI455X (gfx1250) — matrix operand order repaired and checked by reading, not compiled, not run on hardware
//
#include <hip/hip_runtime.h>
#include <math.h>

// ---------------- HGT layer constants ----------------
static constexpr int HIDC = 128;  // hidden
static constexpr int HC   = 8;    // heads
static constexpr int DKC  = 16;   // per-head dim
static constexpr int KNC  = 3;    // node types
static constexpr int KRC  = 3;    // relation types

typedef _Float16 h16v __attribute__((ext_vector_type(16)));
typedef _Float16 h4v  __attribute__((ext_vector_type(4)));
typedef float    f8v  __attribute__((ext_vector_type(8)));

// ================= WMMA fragment loaders (wave32, v_wmma_f32_16x16x32_f16) =================
// A (16x32 f16), per CDNA5 ISA 7.12.2:
//   lanes 0-15 : row M=lane,    elems 0..7 = K kb+0..7,  elems 8..15 = K kb+16..23
//   lanes 16-31: row M=lane-16, elems 0..7 = K kb+8..15, elems 8..15 = K kb+24..31
__device__ __forceinline__ h16v ldsA(const _Float16* xs, int lane, int kb) {
  h16v a;
  const int r  = lane & 15;
  const int ko = (lane & 16) ? 8 : 0;
  const _Float16* p0 = xs + r * HIDC + kb + ko;
#pragma unroll
  for (int i = 0; i < 8; ++i) { a[i] = p0[i]; a[8 + i] = p0[16 + i]; }
  return a;
}
// B fragments are PRE-PACKED in workspace as [t][ktile s][coltile c][lane][16]:
// element i of lane n holds B(K = s*32 + 8*(lane>=16) + (i<8 ? i : 8+i), N = c*16 + (lane&15)).
// One 32-byte contiguous load per fragment.
__device__ __forceinline__ h16v ldBpk(const _Float16* Wpk, int t, int s, int ct, int lane) {
  return *(const h16v*)(Wpk + (((((t * 4 + s) * 8 + ct) * 32) + lane) << 4));
}

// ================= small helpers =================
__global__ void hgt_fill(float* __restrict__ p, float v, int n) {
  int i = blockIdx.x * blockDim.x + threadIdx.x;
  if (i < n) p[i] = v;
}

// Convert f32 weights -> f16 packed directly into the WMMA B-fragment layout.
// gid bits: [3:0]=elem i, [8:4]=lane, [11:9]=coltile, [13:12]=ktile, [15:14]=type
__global__ void hgt_pack_w(const float* __restrict__ a, const float* __restrict__ b,
                           const float* __restrict__ c, const float* __restrict__ d,
                           _Float16* __restrict__ oa, _Float16* __restrict__ ob,
                           _Float16* __restrict__ oc, _Float16* __restrict__ od, int n) {
  int gid = blockIdx.x * blockDim.x + threadIdx.x;
  if (gid >= n) return;
  int i    = gid & 15;
  int lane = (gid >> 4) & 31;
  int ct   = (gid >> 9) & 7;
  int s    = (gid >> 12) & 3;
  int t    = gid >> 14;
  int k    = s * 32 + ((lane & 16) ? 8 : 0) + ((i < 8) ? i : 8 + i);
  int col  = ct * 16 + (lane & 15);
  int src  = t * HIDC * HIDC + k * HIDC + col;
  oa[gid] = (_Float16)a[src];
  ob[gid] = (_Float16)b[src];
  oc[gid] = (_Float16)c[src];
  od[gid] = (_Float16)d[src];
}

__device__ __forceinline__ void atomicMaxF(float* addr, float val) {
  if (val >= 0.0f) atomicMax((int*)addr, __float_as_int(val));
  else             atomicMin((unsigned int*)addr, __float_as_uint(val));
}

// ================= Kernel 1: typed Q/K/V projections via WMMA =================
// 16-node row tile per block; 8 waves, one 16-col output tile per wave.
// Computes x@W[t] for all 3 types, per-row mask-select (matches reference _typed_proj).
__global__ void __launch_bounds__(256)
hgt_qkv(const float* __restrict__ x, const int* __restrict__ ntype,
        const _Float16* __restrict__ Wq, const _Float16* __restrict__ Wk,
        const _Float16* __restrict__ Wv,
        const float* __restrict__ bq, const float* __restrict__ bk,
        const float* __restrict__ bv,
        float* __restrict__ q, float* __restrict__ k, float* __restrict__ v, int N) {
  __shared__ _Float16 xs[16 * HIDC];
  __shared__ int ts[16];
  const int tid  = threadIdx.x;
  const int row0 = blockIdx.x * 16;
  if (row0 + 16 <= N) {  // fast vector path
    for (int i = tid * 4; i < 16 * HIDC; i += 1024) {
      float4 t4 = *(const float4*)(x + row0 * HIDC + i);
      h4v h;
      h[0] = (_Float16)t4.x; h[1] = (_Float16)t4.y;
      h[2] = (_Float16)t4.z; h[3] = (_Float16)t4.w;
      *(h4v*)(xs + i) = h;
    }
  } else {
    const int lim = N * HIDC - 1;
    for (int i = tid; i < 16 * HIDC; i += 256) {
      int gi = row0 * HIDC + i;
      xs[i] = (_Float16)x[gi < lim ? gi : lim];
    }
  }
  if (tid < 16) {
    int rr = row0 + tid;
    ts[tid] = ntype[rr < N ? rr : N - 1];
  }
  __syncthreads();

  const int lane = tid & 31;
  const int ct   = tid >> 5;            // column tile (wave id)
  const int col  = ct * 16 + (lane & 15);

  h16v A[4];
#pragma unroll
  for (int s = 0; s < 4; ++s) A[s] = ldsA(xs, lane, s * 32);

  const _Float16* Ws[3] = {Wq, Wk, Wv};
  const float*    Bb[3] = {bq, bk, bv};
  float*          Oo[3] = {q, k, v};
#pragma unroll
  for (int o = 0; o < 3; ++o) {
    f8v c0 = {}, c1 = {}, c2 = {};
#pragma unroll
    for (int s = 0; s < 4; ++s) {
      h16v b0 = ldBpk(Ws[o], 0, s, ct, lane);
      c0 = __builtin_amdgcn_wmma_f32_16x16x32_f16(false, A[s], false, b0, (short)0, c0, false, false);
      h16v b1 = ldBpk(Ws[o], 1, s, ct, lane);
      c1 = __builtin_amdgcn_wmma_f32_16x16x32_f16(false, A[s], false, b1, (short)0, c1, false, false);
      h16v b2 = ldBpk(Ws[o], 2, s, ct, lane);
      c2 = __builtin_amdgcn_wmma_f32_16x16x32_f16(false, A[s], false, b2, (short)0, c2, false, false);
    }
#pragma unroll
    for (int j = 0; j < 8; ++j) {
      int mr = ((lane & 16) ? 8 : 0) + j;
      int grow = row0 + mr;
      if (grow < N) {
        int t = ts[mr];
        float val = (t == 0) ? c0[j] : (t == 1) ? c1[j] : c2[j];
        val += Bb[o][t * HIDC + col];
        Oo[o][grow * HIDC + col] = val;
      }
    }
  }
}

// ================= Kernel 2: edge attention logits + segment max =================
// One thread per (edge, head). rel_att / rel_pri staged in LDS (24 KB, L2-hot anyway).
__global__ void __launch_bounds__(256)
hgt_att(const float* __restrict__ q, const float* __restrict__ kbuf,
        const int* __restrict__ esrc, const int* __restrict__ edst,
        const int* __restrict__ etype,
        const float* __restrict__ rel_att, const float* __restrict__ rel_pri,
        float* __restrict__ att, float* __restrict__ smax, int E) {
  __shared__ float Ar[KRC * HC * DKC * DKC];
  __shared__ float Pr[KRC * HC];
  const int tid = threadIdx.x;
  for (int i = tid; i < KRC * HC * DKC * DKC; i += 256) Ar[i] = rel_att[i];
  if (tid < KRC * HC) Pr[tid] = rel_pri[tid];
  __syncthreads();

  const int gid = blockIdx.x * 256 + tid;
  if (gid >= E * HC) return;
  const int e = gid >> 3, h = gid & 7;
  const int r = etype[e], s = esrc[e], d = edst[e];
  const float4* qp = (const float4*)(q    + d * HIDC + h * DKC);
  const float4* kp = (const float4*)(kbuf + s * HIDC + h * DKC);
  float qe[DKC], ke[DKC];
#pragma unroll
  for (int i = 0; i < 4; ++i) {
    float4 a4 = qp[i], b4 = kp[i];
    qe[4*i] = a4.x; qe[4*i+1] = a4.y; qe[4*i+2] = a4.z; qe[4*i+3] = a4.w;
    ke[4*i] = b4.x; ke[4*i+1] = b4.y; ke[4*i+2] = b4.z; ke[4*i+3] = b4.w;
  }
  const float* Am = Ar + (r * HC + h) * DKC * DKC;
  float acc = 0.0f;
#pragma unroll
  for (int f = 0; f < DKC; ++f) {
    float kr = 0.0f;
#pragma unroll
    for (int dd = 0; dd < DKC; ++dd) kr += ke[dd] * Am[dd * DKC + f];
    acc += kr * qe[f];
  }
  acc *= Pr[r * HC + h] * 0.25f;  // 1/sqrt(DK=16)
  att[gid] = acc;
  atomicMaxF(&smax[d * HC + h], acc);
}

// ================= Kernel 3: exp + denom =================
__global__ void __launch_bounds__(256)
hgt_expsum(const int* __restrict__ edst, float* __restrict__ att,
           const float* __restrict__ smax, float* __restrict__ den, int E) {
  const int gid = blockIdx.x * 256 + threadIdx.x;
  if (gid >= E * HC) return;
  const int e = gid >> 3, h = gid & 7;
  const int d = edst[e];
  float ex = expf(att[gid] - smax[d * HC + h]);
  att[gid] = ex;
  atomicAdd(&den[d * HC + h], ex);
}

// ================= Kernel 4: msg = v@rel_msg, alpha-weighted scatter-add =================
__global__ void __launch_bounds__(256)
hgt_aggr(const float* __restrict__ vbuf, const int* __restrict__ esrc,
         const int* __restrict__ edst, const int* __restrict__ etype,
         const float* __restrict__ rel_msg, const float* __restrict__ att,
         const float* __restrict__ den, float* __restrict__ aggr, int E) {
  __shared__ float Mr[KRC * HC * DKC * DKC];
  const int tid = threadIdx.x;
  for (int i = tid; i < KRC * HC * DKC * DKC; i += 256) Mr[i] = rel_msg[i];
  __syncthreads();

  const int gid = blockIdx.x * 256 + tid;
  if (gid >= E * HC) return;
  const int e = gid >> 3, h = gid & 7;
  const int r = etype[e], s = esrc[e], d = edst[e];
  const float alpha = att[gid] / (den[d * HC + h] + 1e-16f);
  const float4* vp = (const float4*)(vbuf + s * HIDC + h * DKC);
  float ve[DKC];
#pragma unroll
  for (int i = 0; i < 4; ++i) {
    float4 a4 = vp[i];
    ve[4*i] = a4.x; ve[4*i+1] = a4.y; ve[4*i+2] = a4.z; ve[4*i+3] = a4.w;
  }
  const float* Mm = Mr + (r * HC + h) * DKC * DKC;
  float* ap = aggr + d * HIDC + h * DKC;
#pragma unroll
  for (int f = 0; f < DKC; ++f) {
    float m = 0.0f;
#pragma unroll
    for (int dd = 0; dd < DKC; ++dd) m += ve[dd] * Mm[dd * DKC + f];
    atomicAdd(&ap[f], alpha * m);
  }
}

// ================= Kernel 5: gelu -> typed linear (WMMA) -> residual -> typed layernorm =================
__global__ void __launch_bounds__(256)
hgt_update(const float* __restrict__ x, const float* __restrict__ aggr,
           const int* __restrict__ ntype, const _Float16* __restrict__ Wu,
           const float* __restrict__ bu, const float* __restrict__ gamma,
           const float* __restrict__ beta, float* __restrict__ out, int N) {
  __shared__ _Float16 gs[16 * HIDC];
  __shared__ float    hn[16 * HIDC];
  __shared__ int   ts[16];
  __shared__ float mu_s[16], rs_s[16];
  const int tid  = threadIdx.x;
  const int row0 = blockIdx.x * 16;
  const int lim  = N * HIDC - 1;
  if (row0 + 16 <= N) {  // fast vector path
    for (int i = tid * 4; i < 16 * HIDC; i += 1024) {
      float4 t4 = *(const float4*)(aggr + row0 * HIDC + i);
      h4v h;
      h[0] = (_Float16)(0.5f * t4.x * (1.0f + erff(t4.x * 0.70710678118654752f)));
      h[1] = (_Float16)(0.5f * t4.y * (1.0f + erff(t4.y * 0.70710678118654752f)));
      h[2] = (_Float16)(0.5f * t4.z * (1.0f + erff(t4.z * 0.70710678118654752f)));
      h[3] = (_Float16)(0.5f * t4.w * (1.0f + erff(t4.w * 0.70710678118654752f)));
      *(h4v*)(gs + i) = h;
    }
  } else {
    for (int i = tid; i < 16 * HIDC; i += 256) {
      int gi = row0 * HIDC + i;
      float a = aggr[gi < lim ? gi : lim];
      gs[i] = (_Float16)(0.5f * a * (1.0f + erff(a * 0.70710678118654752f)));
    }
  }
  if (tid < 16) {
    int rr = row0 + tid;
    ts[tid] = ntype[rr < N ? rr : N - 1];
  }
  __syncthreads();

  const int lane = tid & 31;
  const int ct   = tid >> 5;
  const int col  = ct * 16 + (lane & 15);

  f8v c0 = {}, c1 = {}, c2 = {};
#pragma unroll
  for (int s = 0; s < 4; ++s) {
    h16v A  = ldsA(gs, lane, s * 32);
    h16v b0 = ldBpk(Wu, 0, s, ct, lane);
    c0 = __builtin_amdgcn_wmma_f32_16x16x32_f16(false, A, false, b0, (short)0, c0, false, false);
    h16v b1 = ldBpk(Wu, 1, s, ct, lane);
    c1 = __builtin_amdgcn_wmma_f32_16x16x32_f16(false, A, false, b1, (short)0, c1, false, false);
    h16v b2 = ldBpk(Wu, 2, s, ct, lane);
    c2 = __builtin_amdgcn_wmma_f32_16x16x32_f16(false, A, false, b2, (short)0, c2, false, false);
  }
#pragma unroll
  for (int j = 0; j < 8; ++j) {
    int mr = ((lane & 16) ? 8 : 0) + j;
    int grow = row0 + mr;
    int t = ts[mr];
    float val = (t == 0) ? c0[j] : (t == 1) ? c1[j] : c2[j];
    val += bu[t * HIDC + col];
    int gi = grow * HIDC + col;
    val += x[gi < lim ? gi : lim];  // residual
    hn[mr * HIDC + col] = val;
  }
  __syncthreads();

  if (tid < 16) {
    float s = 0.0f, ss = 0.0f;
    const float* rp = hn + tid * HIDC;
#pragma unroll 8
    for (int c = 0; c < HIDC; ++c) { float v = rp[c]; s += v; ss += v * v; }
    float mu  = s * (1.0f / HIDC);
    float var = ss * (1.0f / HIDC) - mu * mu;
    mu_s[tid] = mu;
    rs_s[tid] = rsqrtf(var + 1e-5f);
  }
  __syncthreads();

  for (int i = tid; i < 16 * HIDC; i += 256) {
    int r = i >> 7, c = i & 127;
    int grow = row0 + r;
    if (grow < N) {
      int t = ts[r];
      float v = (hn[i] - mu_s[r]) * rs_s[r];
      out[grow * HIDC + c] = v * gamma[t * HIDC + c] + beta[t * HIDC + c];
    }
  }
}

// ================= launch =================
extern "C" void kernel_launch(void* const* d_in, const int* in_sizes, int n_in,
                              void* d_out, int out_size, void* d_ws, size_t ws_size,
                              hipStream_t stream) {
  const float* x       = (const float*)d_in[0];
  const float* Wq      = (const float*)d_in[1];
  const float* bq      = (const float*)d_in[2];
  const float* Wk      = (const float*)d_in[3];
  const float* bk      = (const float*)d_in[4];
  const float* Wv      = (const float*)d_in[5];
  const float* bv      = (const float*)d_in[6];
  const float* rel_pri = (const float*)d_in[7];
  const float* rel_att = (const float*)d_in[8];
  const float* rel_msg = (const float*)d_in[9];
  const float* Wu      = (const float*)d_in[10];
  const float* bu      = (const float*)d_in[11];
  const float* gamma   = (const float*)d_in[12];
  const float* beta    = (const float*)d_in[13];
  const int* esrc      = (const int*)d_in[14];
  const int* edst      = (const int*)d_in[15];
  const int* ntype     = (const int*)d_in[16];
  const int* etype     = (const int*)d_in[17];
  const int N = in_sizes[16];
  const int E = in_sizes[14];

  // workspace carve-out (~122 MB)
  char* w = (char*)d_ws;
  auto carve = [&](size_t bytes) -> void* {
    void* p = (void*)w;
    w += (bytes + 255) & ~(size_t)255;
    return p;
  };
  float* q    = (float*)carve((size_t)N * HIDC * 4);
  float* kk   = (float*)carve((size_t)N * HIDC * 4);
  float* vv   = (float*)carve((size_t)N * HIDC * 4);
  float* att  = (float*)carve((size_t)E * HC * 4);
  float* smax = (float*)carve((size_t)N * HC * 4);
  float* den  = (float*)carve((size_t)N * HC * 4);
  float* aggr = (float*)carve((size_t)N * HIDC * 4);
  _Float16* wq16 = (_Float16*)carve((size_t)KNC * HIDC * HIDC * 2);
  _Float16* wk16 = (_Float16*)carve((size_t)KNC * HIDC * HIDC * 2);
  _Float16* wv16 = (_Float16*)carve((size_t)KNC * HIDC * HIDC * 2);
  _Float16* wu16 = (_Float16*)carve((size_t)KNC * HIDC * HIDC * 2);

  const int wn = KNC * HIDC * HIDC;
  hgt_pack_w<<<(wn + 255) / 256, 256, 0, stream>>>(Wq, Wk, Wv, Wu, wq16, wk16, wv16, wu16, wn);

  hgt_fill<<<((N * HIDC) + 255) / 256, 256, 0, stream>>>(aggr, 0.0f, N * HIDC);
  hgt_fill<<<((N * HC) + 255) / 256, 256, 0, stream>>>(den, 0.0f, N * HC);
  hgt_fill<<<((N * HC) + 255) / 256, 256, 0, stream>>>(smax, -3.0e38f, N * HC);

  hgt_qkv<<<(N + 15) / 16, 256, 0, stream>>>(x, ntype, wq16, wk16, wv16,
                                             bq, bk, bv, q, kk, vv, N);

  const int EH = E * HC;
  hgt_att<<<(EH + 255) / 256, 256, 0, stream>>>(q, kk, esrc, edst, etype,
                                                rel_att, rel_pri, att, smax, E);
  hgt_expsum<<<(EH + 255) / 256, 256, 0, stream>>>(edst, att, smax, den, E);
  hgt_aggr<<<(EH + 255) / 256, 256, 0, stream>>>(vv, esrc, edst, etype,
                                                 rel_msg, att, den, aggr, E);

  hgt_update<<<(N + 15) / 16, 256, 0, stream>>>(x, aggr, ntype, wu16, bu,
                                                gamma, beta, (float*)d_out, N);
}
